// QuantOPTAttention_28226525069368
// MI455X (gfx1250) — 8-bit matrix operand order repaired and checked by reading, compiled, not run on hardware
//
#include <hip/hip_runtime.h>
#include <stdint.h>

#define BB 2
#define TT 1024
#define EE 2048
#define HH 32
#define DD 64
#define NN (BB*TT)
#define KK EE

typedef int v8i __attribute__((ext_vector_type(8)));

__device__ __forceinline__ uint32_t f2key(float f) {
  uint32_t u = __float_as_uint(f);
  return (u & 0x80000000u) ? ~u : (u | 0x80000000u);
}
__device__ __forceinline__ float key2f(uint32_t k) {
  uint32_t u = (k & 0x80000000u) ? (k & 0x7fffffffu) : ~k;
  return __uint_as_float(u);
}

__device__ __forceinline__ void load_scale_zp(const uint32_t* mm, int slot,
                                              float& scale, int& zp) {
  float xmin = fminf(key2f(mm[2*slot+0]), 0.0f);
  float xmax = fmaxf(key2f(mm[2*slot+1]), 0.0f);
  scale = fmaxf((xmax - xmin) * (1.0f/255.0f), 1e-8f);
  zp = (int)rintf(-xmin / scale);
}

__global__ void init_mm_kernel(uint32_t* mm) {
  int i = threadIdx.x;
  if (i < 16) { mm[2*i] = 0xffffffffu; mm[2*i+1] = 0u; }
}

__global__ __launch_bounds__(256) void minmax_kernel(const float* __restrict__ src,
                                                     int n, uint32_t* mm, int slot) {
  float lo = 3.402823466e38f, hi = -3.402823466e38f;
  for (int i = blockIdx.x*256 + threadIdx.x; i < n; i += gridDim.x*256) {
    float v = src[i];
    lo = fminf(lo, v); hi = fmaxf(hi, v);
  }
  uint32_t klo = f2key(lo), khi = f2key(hi);
  for (int off = 16; off; off >>= 1) {
    klo = min(klo, (uint32_t)__shfl_xor((int)klo, off, 32));
    khi = max(khi, (uint32_t)__shfl_xor((int)khi, off, 32));
  }
  if ((threadIdx.x & 31) == 0) {
    atomicMin(&mm[2*slot],   klo);
    atomicMax(&mm[2*slot+1], khi);
  }
}

__global__ __launch_bounds__(256) void quant_kernel(const float* __restrict__ src,
    uint8_t* __restrict__ dst, int* __restrict__ sums, const uint32_t* mm, int slot,
    int rowLen, int groupSize, int transT) {
  int row = blockIdx.x, tid = threadIdx.x;
  float scale; int zp;
  load_scale_zp(mm, slot, scale, zp);
  float inv = 1.0f / scale;
  __shared__ int part[256];
  const float* sp = src + (size_t)row * rowLen;
  int per = rowLen / 256;
  int e0 = tid * per;
  int qv[8]; int local = 0;
  for (int i = 0; i < per; ++i) {
    float q = rintf(sp[e0+i] * inv) + (float)zp;
    q = fminf(fmaxf(q, 0.0f), 255.0f);
    int qi = (int)q;
    qv[i] = qi; local += qi;
  }
  if (transT > 0) {
    int b = row / transT, t = row % transT;
    size_t base = (size_t)b * rowLen * transT + t;
    for (int i = 0; i < per; ++i) dst[base + (size_t)(e0+i) * transT] = (uint8_t)qv[i];
  } else {
    uint8_t* dp = dst + (size_t)row * rowLen + e0;
    for (int i = 0; i < per; ++i) dp[i] = (uint8_t)qv[i];
  }
  if (sums) {
    part[tid] = local;
    __syncthreads();
    int numGroups = rowLen / groupSize;
    int tpg = 256 / numGroups;
    if (tid < numGroups) {
      int s = 0;
      for (int i = 0; i < tpg; ++i) s += part[tid*tpg + i];
      sums[(size_t)row * numGroups + tid] = s;
    }
  }
}

__global__ __launch_bounds__(256) void gemm_kernel(const uint8_t* __restrict__ A,
    const uint8_t* __restrict__ Bm, const int* __restrict__ asum,
    const int* __restrict__ bsum, const float* __restrict__ bias,
    const uint32_t* mm, int slotA, int slotB, float extraScale,
    float* __restrict__ Cout, uint32_t* mmOut, int slotOut) {
  __shared__ __align__(16) uint8_t As[128*64];
  __shared__ __align__(16) uint8_t Bs[128*64];
  int tid = threadIdx.x;
  int wid = tid >> 5, lane = tid & 31;
  int m = lane & 15, hif = lane >> 4;
  int wr = wid >> 2, wc = wid & 3;
  int n0 = blockIdx.y * 128, m0 = blockIdx.x * 128;
  v8i zero = {};
  v8i acc[4][2];
  for (int i = 0; i < 4; ++i) for (int j = 0; j < 2; ++j) acc[i][j] = zero;

  int ldRow = tid >> 1, ldOff = (tid & 1) * 32;
  for (int kk = 0; kk < KK; kk += 64) {
    const uint4* ga = (const uint4*)(A  + (size_t)(n0 + ldRow)*KK + kk + ldOff);
    const uint4* gb = (const uint4*)(Bm + (size_t)(m0 + ldRow)*KK + kk + ldOff);
    uint4 a0 = ga[0], a1 = ga[1], b0 = gb[0], b1 = gb[1];
    __syncthreads();
    *(uint4*)(As + ldRow*64 + ldOff)      = a0;
    *(uint4*)(As + ldRow*64 + ldOff + 16) = a1;
    *(uint4*)(Bs + ldRow*64 + ldOff)      = b0;
    *(uint4*)(Bs + ldRow*64 + ldOff + 16) = b1;
    __syncthreads();

    v8i af[4], bf[2];
    for (int i = 0; i < 4; ++i) {
      const uint8_t* p = As + (wr*64 + i*16 + m)*64 + hif*16;
      uint4 d0 = *(const uint4*)(p);
      uint4 d1 = *(const uint4*)(p + 32);
      af[i][0]=(int)d0.x; af[i][1]=(int)d0.y; af[i][2]=(int)d0.z; af[i][3]=(int)d0.w;
      af[i][4]=(int)d1.x; af[i][5]=(int)d1.y; af[i][6]=(int)d1.z; af[i][7]=(int)d1.w;
    }
    for (int j = 0; j < 2; ++j) {
      const uint8_t* p = Bs + (wc*32 + j*16 + m)*64 + hif*16;
      uint4 lo = *(const uint4*)(p);
      uint4 hq = *(const uint4*)(p + 32);
      bf[j][0]=(int)lo.x; bf[j][1]=(int)lo.y; bf[j][2]=(int)lo.z; bf[j][3]=(int)lo.w;
      bf[j][4]=(int)hq.x; bf[j][5]=(int)hq.y; bf[j][6]=(int)hq.z; bf[j][7]=(int)hq.w;
    }
    for (int i = 0; i < 4; ++i)
      for (int j = 0; j < 2; ++j)
        acc[i][j] = __builtin_amdgcn_wmma_i32_16x16x64_iu8(
            false, af[i], false, bf[j], acc[i][j], false, false);
  }

  float sA, sB; int zA, zB;
  load_scale_zp(mm, slotA, sA, zA);
  load_scale_zp(mm, slotB, sB, zB);
  float ss = sA * sB;
  int kzz = KK * zA * zB;
  float lmin = 3.402823466e38f, lmax = -3.402823466e38f;
  for (int i = 0; i < 4; ++i) {
    int rbase = n0 + wr*64 + i*16 + hif*8;
    int asv[8];
    for (int g = 0; g < 8; ++g) asv[g] = asum[rbase + g];
    for (int j = 0; j < 2; ++j) {
      int c = m0 + wc*32 + j*16 + m;
      int bs = bsum[c];
      float bv = bias[c];
      for (int g = 0; g < 8; ++g) {
        int corr = acc[i][j][g] - zA*bs - zB*asv[g] + kzz;
        float y = ((float)corr * ss + bv) * extraScale;
        Cout[(size_t)(rbase + g) * EE + c] = y;
        lmin = fminf(lmin, y); lmax = fmaxf(lmax, y);
      }
    }
  }
  if (mmOut) {
    atomicMin(&mmOut[2*slotOut],   f2key(lmin));
    atomicMax(&mmOut[2*slotOut+1], f2key(lmax));
  }
}

__global__ __launch_bounds__(256) void attn_kernel(const uint8_t* __restrict__ qq,
    const uint8_t* __restrict__ kq, const uint8_t* __restrict__ vqt,
    const int* __restrict__ qsum, const int* __restrict__ ksum,
    const float* __restrict__ mask, const uint32_t* mm,
    float* __restrict__ attn, uint32_t* mmOut) {
  extern __shared__ __align__(16) char smem[];
  float*   sc  = (float*)smem;
  uint8_t* pq  = (uint8_t*)(smem + 16*1024*4);
  float*   red = (float*)(smem + 16*1024*4 + 16*1024);
  int*     psh = (int*)(smem + 16*1024*4 + 16*1024 + 16*16*4);

  int tid = threadIdx.x, wid = tid >> 5, lane = tid & 31;
  int m = lane & 15, hif = lane >> 4;
  int bh = blockIdx.y; int b = bh >> 5, h = bh & 31;
  int t0 = blockIdx.x * 16;
  if (tid < 16) psh[tid] = 0;

  float sQ, sK, sV; int zQ, zK, zV;
  load_scale_zp(mm, 5, sQ, zQ);
  load_scale_zp(mm, 6, sK, zK);
  load_scale_zp(mm, 7, sV, zV);

  {
    const uint8_t* p = qq + (size_t)b*TT*EE + (size_t)(t0 + m)*EE + h*DD + hif*16;
    uint4 d0 = *(const uint4*)(p);
    uint4 d1 = *(const uint4*)(p + 32);
    v8i afrag;
    afrag[0]=(int)d0.x; afrag[1]=(int)d0.y; afrag[2]=(int)d0.z; afrag[3]=(int)d0.w;
    afrag[4]=(int)d1.x; afrag[5]=(int)d1.y; afrag[6]=(int)d1.z; afrag[7]=(int)d1.w;
    int qsv[8];
    for (int g = 0; g < 8; ++g)
      qsv[g] = qsum[((size_t)b*TT + t0 + g + hif*8)*HH + h];
    float ss = sQ * sK;
    int dzz = DD * zQ * zK;
    v8i zero = {};
    for (int j = wid; j < TT/16; j += 8) {
      int s0 = j * 16;
      const uint8_t* bp = kq + (size_t)b*TT*EE + (size_t)(s0 + m)*EE + h*DD + hif*16;
      uint4 lo = *(const uint4*)(bp);
      uint4 hq = *(const uint4*)(bp + 32);
      v8i bfrag;
      bfrag[0]=(int)lo.x; bfrag[1]=(int)lo.y; bfrag[2]=(int)lo.z; bfrag[3]=(int)lo.w;
      bfrag[4]=(int)hq.x; bfrag[5]=(int)hq.y; bfrag[6]=(int)hq.z; bfrag[7]=(int)hq.w;
      v8i acc = __builtin_amdgcn_wmma_i32_16x16x64_iu8(
          false, afrag, false, bfrag, zero, false, false);
      int s = s0 + m;
      int ksv = ksum[((size_t)b*TT + s)*HH + h];
      for (int g = 0; g < 8; ++g) {
        int r = g + hif*8;
        int corr = acc[g] - zQ*ksv - zK*qsv[g] + dzz;
        float sco = (float)corr * ss;
        float mk = mask[(size_t)b*TT*TT + (size_t)(t0 + r)*TT + s];
        sco = fmaxf(sco + mk, -3.402823466e38f);
        sc[r*TT + s] = sco;
      }
    }
  }
  __syncthreads();

  {
    int row = tid >> 4, i = tid & 15;
    float mx = -3.402823466e38f;
    for (int jj = 0; jj < 64; ++jj) mx = fmaxf(mx, sc[row*TT + i + jj*16]);
    red[row*16 + i] = mx;
    __syncthreads();
    float rmax = -3.402823466e38f;
    for (int t = 0; t < 16; ++t) rmax = fmaxf(rmax, red[row*16 + t]);
    __syncthreads();
    float sm = 0.0f;
    for (int jj = 0; jj < 64; ++jj) {
      int e = i + jj*16;
      float ex = expf(sc[row*TT + e] - rmax);
      sc[row*TT + e] = ex;
      sm += ex;
    }
    red[row*16 + i] = sm;
    __syncthreads();
    float rsum = 0.0f;
    for (int t = 0; t < 16; ++t) rsum += red[row*16 + t];
    float inv = 1.0f / rsum;
    int local = 0;
    for (int jj = 0; jj < 64; ++jj) {
      int e = i + jj*16;
      float p = sc[row*TT + e] * inv;
      float q = fminf(fmaxf(rintf(p * 255.0f), 0.0f), 255.0f);
      int qi = (int)q;
      pq[row*TT + e] = (uint8_t)qi;
      local += qi;
    }
    atomicAdd(&psh[row], local);
  }
  __syncthreads();

  if (wid < 4) {
    int dt = wid;
    v8i acc = {};
    for (int kkk = 0; kkk < TT; kkk += 64) {
      const uint8_t* ap = pq + m*TT + kkk + hif*16;
      uint4 d0 = *(const uint4*)(ap);
      uint4 d1 = *(const uint4*)(ap + 32);
      v8i af;
      af[0]=(int)d0.x; af[1]=(int)d0.y; af[2]=(int)d0.z; af[3]=(int)d0.w;
      af[4]=(int)d1.x; af[5]=(int)d1.y; af[6]=(int)d1.z; af[7]=(int)d1.w;
      const uint8_t* bp = vqt + (size_t)b*EE*TT + (size_t)(h*DD + dt*16 + m)*TT + kkk + hif*16;
      uint4 lo = *(const uint4*)(bp);
      uint4 hq = *(const uint4*)(bp + 32);
      v8i bf;
      bf[0]=(int)lo.x; bf[1]=(int)lo.y; bf[2]=(int)lo.z; bf[3]=(int)lo.w;
      bf[4]=(int)hq.x; bf[5]=(int)hq.y; bf[6]=(int)hq.z; bf[7]=(int)hq.w;
      acc = __builtin_amdgcn_wmma_i32_16x16x64_iu8(false, af, false, bf, acc, false, false);
    }
    float ss = sV * (1.0f/255.0f);
    float lmin = 3.402823466e38f, lmax = -3.402823466e38f;
    for (int g = 0; g < 8; ++g) {
      int r = g + hif*8;
      int corr = acc[g] - zV * psh[r];
      float y = (float)corr * ss;
      attn[(size_t)b*TT*EE + (size_t)(t0 + r)*EE + h*DD + dt*16 + m] = y;
      lmin = fminf(lmin, y); lmax = fmaxf(lmax, y);
    }
    atomicMin(&mmOut[2*8],   f2key(lmin));
    atomicMax(&mmOut[2*8+1], f2key(lmax));
  }
}

extern "C" void kernel_launch(void* const* d_in, const int* in_sizes, int n_in,
                              void* d_out, int out_size, void* d_ws, size_t ws_size,
                              hipStream_t stream) {
  (void)in_sizes; (void)n_in; (void)out_size; (void)ws_size;
  const float* x    = (const float*)d_in[0];
  const float* mask = (const float*)d_in[1];
  const float* Wq   = (const float*)d_in[2];
  const float* bq   = (const float*)d_in[3];
  const float* Wk   = (const float*)d_in[4];
  const float* bk   = (const float*)d_in[5];
  const float* Wv   = (const float*)d_in[6];
  const float* bv   = (const float*)d_in[7];
  const float* Wo   = (const float*)d_in[8];
  const float* bo   = (const float*)d_in[9];

  char* ws = (char*)d_ws;
  size_t off = 0;
  auto alloc = [&](size_t bytes) -> void* {
    void* p = (void*)(ws + off);
    off += (bytes + 255) & ~(size_t)255;
    return p;
  };
  uint32_t* mm   = (uint32_t*)alloc(256);
  uint8_t* xq    = (uint8_t*)alloc((size_t)NN*KK);
  uint8_t* wqq   = (uint8_t*)alloc((size_t)EE*KK);
  uint8_t* wkq   = (uint8_t*)alloc((size_t)EE*KK);
  uint8_t* wvq   = (uint8_t*)alloc((size_t)EE*KK);
  uint8_t* woq   = (uint8_t*)alloc((size_t)EE*KK);
  uint8_t* attq  = (uint8_t*)alloc((size_t)NN*EE);
  uint8_t* qqb   = (uint8_t*)alloc((size_t)NN*EE);
  uint8_t* kqb   = (uint8_t*)alloc((size_t)NN*EE);
  uint8_t* vqt   = (uint8_t*)alloc((size_t)BB*EE*TT);
  int* xsum      = (int*)alloc((size_t)NN*4);
  int* wqsum     = (int*)alloc((size_t)EE*4);
  int* wksum     = (int*)alloc((size_t)EE*4);
  int* wvsum     = (int*)alloc((size_t)EE*4);
  int* wosum     = (int*)alloc((size_t)EE*4);
  int* attsum    = (int*)alloc((size_t)NN*4);
  int* qsumb     = (int*)alloc((size_t)NN*HH*4);
  int* ksumb     = (int*)alloc((size_t)NN*HH*4);
  float* qf      = (float*)alloc((size_t)NN*EE*4);
  float* kf      = (float*)alloc((size_t)NN*EE*4);
  float* vf      = (float*)alloc((size_t)NN*EE*4);
  float* attf    = (float*)alloc((size_t)NN*EE*4);

  init_mm_kernel<<<1, 32, 0, stream>>>(mm);

  minmax_kernel<<<1024, 256, 0, stream>>>(x,  NN*EE, mm, 0);
  minmax_kernel<<<1024, 256, 0, stream>>>(Wq, EE*EE, mm, 1);
  minmax_kernel<<<1024, 256, 0, stream>>>(Wk, EE*EE, mm, 2);
  minmax_kernel<<<1024, 256, 0, stream>>>(Wv, EE*EE, mm, 3);
  minmax_kernel<<<1024, 256, 0, stream>>>(Wo, EE*EE, mm, 4);

  quant_kernel<<<NN, 256, 0, stream>>>(x,  xq,  xsum,  mm, 0, KK, KK, 0);
  quant_kernel<<<EE, 256, 0, stream>>>(Wq, wqq, wqsum, mm, 1, KK, KK, 0);
  quant_kernel<<<EE, 256, 0, stream>>>(Wk, wkq, wksum, mm, 2, KK, KK, 0);
  quant_kernel<<<EE, 256, 0, stream>>>(Wv, wvq, wvsum, mm, 3, KK, KK, 0);
  quant_kernel<<<EE, 256, 0, stream>>>(Wo, woq, wosum, mm, 4, KK, KK, 0);

  dim3 gg(EE/128, NN/128);
  gemm_kernel<<<gg, 256, 0, stream>>>(xq, wqq, xsum, wqsum, bq, mm, 0, 1, 0.125f, qf, mm, 5);
  gemm_kernel<<<gg, 256, 0, stream>>>(xq, wkq, xsum, wksum, bk, mm, 0, 2, 1.0f,  kf, mm, 6);
  gemm_kernel<<<gg, 256, 0, stream>>>(xq, wvq, xsum, wvsum, bv, mm, 0, 3, 1.0f,  vf, mm, 7);

  quant_kernel<<<NN, 256, 0, stream>>>(qf, qqb, qsumb, mm, 5, EE, DD, 0);
  quant_kernel<<<NN, 256, 0, stream>>>(kf, kqb, ksumb, mm, 6, EE, DD, 0);
  quant_kernel<<<NN, 256, 0, stream>>>(vf, vqt, nullptr, mm, 7, EE, KK, TT);

  size_t shbytes = 16*1024*4 + 16*1024 + 16*16*4 + 64;
  attn_kernel<<<dim3(TT/16, BB*HH), 256, shbytes, stream>>>(
      qqb, kqb, vqt, qsumb, ksumb, mask, mm, attf, mm);

  quant_kernel<<<NN, 256, 0, stream>>>(attf, attq, attsum, mm, 8, EE, EE, 0);

  gemm_kernel<<<gg, 256, 0, stream>>>(attq, woq, attsum, wosum, bo, mm, 8, 4, 1.0f,
                                      (float*)d_out, nullptr, 0);
}
